// MultiHeadAttention_17867063951632
// MI455X (gfx1250) — matrix operand order repaired and checked by reading, not compiled, not run on hardware
//
#include <hip/hip_runtime.h>
#include <hip/hip_bf16.h>
#include <stdint.h>

// ---------------------------------------------------------------------------
// Types for CDNA5 WMMA (wave32)
// ---------------------------------------------------------------------------
typedef __attribute__((ext_vector_type(16))) __bf16 v16bf;
typedef __attribute__((ext_vector_type(8)))  float  v8f;
typedef __attribute__((ext_vector_type(4)))  unsigned int u32x4;
typedef __attribute__((ext_vector_type(4)))  int    i32x4;
typedef __attribute__((ext_vector_type(8)))  int    i32x8;

union ABf { v16bf v; uint4 u[2]; };   // 16 bf16 = 32B per lane (A or B operand)
union CF  { v8f  v; float f[8]; };    // 16x16 f32 C/D tile: 8 regs per lane

__device__ __forceinline__ uint16_t f2bf(float x) {
    union { float f; uint32_t u; } c; c.f = x;
    uint32_t u = c.u;
    return (uint16_t)((u + 0x7FFFu + ((u >> 16) & 1u)) >> 16);   // RNE
}
__device__ __forceinline__ float bf2f(uint16_t h) {
    union { uint32_t u; float f; } c; c.u = ((uint32_t)h) << 16;
    return c.f;
}

#define WMMA_BF16(A, B, C) \
    __builtin_amdgcn_wmma_f32_16x16x32_bf16(false, (A), false, (B), (short)0, (C), false, false)

// Issue a TDM 2D tile load: contiguous-row tile of bf16, rows x cols elements.
// ldsb     : 32-bit LDS byte address of destination
// gaddr    : 64-bit global byte address of tile start
// rows,cols: tile dims in elements (cols = row length, row stride = cols)
__device__ __forceinline__ void tdm_load_tile_bf16(uint32_t ldsb, uint64_t gaddr,
                                                   uint32_t rows, uint32_t cols) {
    u32x4 g0;
    g0.x = 1u;                                              // count=1, user mode
    g0.y = ldsb;                                            // lds_addr
    g0.z = (uint32_t)(gaddr & 0xFFFFFFFFu);                 // global_addr[31:0]
    g0.w = (uint32_t)((gaddr >> 32) & 0x1FFFFFFu) | (2u << 30);  // [56:32] | type=2
    i32x8 g1;
    g1[0] = 0x00010000;                                     // data_size=1 (2 bytes)
    g1[1] = (int)(cols << 16);                              // tensor_dim0
    g1[2] = (int)(rows << 16);                              // tensor_dim1
    g1[3] = (int)(cols << 16);                              // tile_dim0
    g1[4] = (int)rows;                                      // tile_dim1
    g1[5] = (int)cols;                                      // tensor_dim0_stride
    g1[6] = 0;
    g1[7] = 0;
    i32x4 z4 = {0, 0, 0, 0};
#if defined(__clang_major__) && __clang_major__ >= 23
    i32x8 z8 = {0, 0, 0, 0, 0, 0, 0, 0};
    __builtin_amdgcn_tensor_load_to_lds(g0, g1, z4, z4, z8, 0);
#else
    __builtin_amdgcn_tensor_load_to_lds(g0, g1, z4, z4, 0);
#endif
}

// Async global->LDS 16-byte copy (GVS addressing), tracked by ASYNCcnt.
__device__ __forceinline__ void async_ld_b128(uint32_t lds_addr, uint32_t byte_off,
                                              const void* base) {
    asm volatile("global_load_async_to_lds_b128 %0, %1, %2"
                 :
                 : "v"(lds_addr), "v"(byte_off), "s"(base)
                 : "memory");
}

// ---------------------------------------------------------------------------
// Generic GEMM: C[M,N] = A[M,K] @ W[N,K]^T  (fp32 in, bf16 WMMA compute)
// Epilogues:
//   outF  : row-major fp32 [M,N], + biasF[n]                (final projection)
//   out0/1: head-major bf16 [b,h,seq,64], + bias0/1[n]      (q/k/v/r paths)
// Tile: 64x64 per workgroup (4 waves, each a 16-row strip), K-step 32.
// ---------------------------------------------------------------------------
__global__ void __launch_bounds__(128)
gemm_bf16_wmma(const float* __restrict__ A, const float* __restrict__ W,
               int M, int N, int K,
               float* __restrict__ outF, const float* __restrict__ biasF,
               uint16_t* __restrict__ out0, const float* __restrict__ bias0,
               uint16_t* __restrict__ out1, const float* __restrict__ bias1,
               int seq)
{
    __shared__ uint16_t sA[64 * 32];
    __shared__ uint16_t sB[64 * 32];

    const int t    = threadIdx.x;
    const int w    = t >> 5;
    const int lane = t & 31;
    const int half = lane >> 4;
    const int l15  = lane & 15;
    const int m0   = blockIdx.x * 64;
    const int n0   = blockIdx.y * 64;

    CF acc[4];
#pragma unroll
    for (int nt = 0; nt < 4; ++nt)
#pragma unroll
        for (int v = 0; v < 8; ++v) acc[nt].f[v] = 0.f;

    for (int k0 = 0; k0 < K; k0 += 32) {
        // prefetch next K-step tiles into cache (global_prefetch_b8)
        if (k0 + 32 < K) {
            __builtin_prefetch(A + (size_t)(m0 + (t >> 1)) * K + k0 + 32 + (t & 1) * 16, 0, 1);
            __builtin_prefetch(W + (size_t)(n0 + (t >> 1)) * K + k0 + 32 + (t & 1) * 16, 0, 1);
        }
        // ---- stage 64x32 of A and W into LDS as bf16 ----
#pragma unroll
        for (int it = 0; it < 4; ++it) {
            int idx = t * 16 + it * 4;          // < 2048
            int row = idx >> 5, col = idx & 31;
            float4 av = *(const float4*)(A + (size_t)(m0 + row) * K + k0 + col);
            uint2 pa;
            pa.x = (uint32_t)f2bf(av.x) | ((uint32_t)f2bf(av.y) << 16);
            pa.y = (uint32_t)f2bf(av.z) | ((uint32_t)f2bf(av.w) << 16);
            *(uint2*)(sA + row * 32 + col) = pa;
            float4 bv = *(const float4*)(W + (size_t)(n0 + row) * K + k0 + col);
            uint2 pb;
            pb.x = (uint32_t)f2bf(bv.x) | ((uint32_t)f2bf(bv.y) << 16);
            pb.y = (uint32_t)f2bf(bv.z) | ((uint32_t)f2bf(bv.w) << 16);
            *(uint2*)(sB + row * 32 + col) = pb;
        }
        __syncthreads();

        // A strip for this wave (16x32), CDNA5 A layout:
        // lane<16: M=lane, K{0..7,16..23}; lane>=16: M=lane-16, K{8..15,24..31}
        ABf a;
        const uint4* pa = (const uint4*)(sA + (w * 16 + l15) * 32);
        a.u[0] = pa[half];
        a.u[1] = pa[2 + half];

#pragma unroll
        for (int nt = 0; nt < 4; ++nt) {
            // B tile 32x16: lane<16: col=lane, K{0..7,16..23} ; lane>=16: col, K{8..15,24..31}
            ABf b;
            const uint4* pb = (const uint4*)(sB + (nt * 16 + l15) * 32);
            b.u[0] = pb[half];
            b.u[1] = pb[2 + half];
            acc[nt].v = WMMA_BF16(a.v, b.v, acc[nt].v);
        }
        __syncthreads();
    }

    // ---- epilogue ----
    const int HN = N >> 6;
#pragma unroll
    for (int nt = 0; nt < 4; ++nt)
#pragma unroll
        for (int v = 0; v < 8; ++v) {
            int m = m0 + w * 16 + v + 8 * half;
            int n = n0 + nt * 16 + l15;
            float c = acc[nt].f[v];
            if (outF) outF[(size_t)m * N + n] = c + (biasF ? biasF[n] : 0.f);
            if (out0) {
                int bb = m / seq, ii = m % seq, hh = n >> 6, dd = n & 63;
                size_t o = (((size_t)bb * HN + hh) * seq + ii) * 64 + dd;
                out0[o] = f2bf(c + (bias0 ? bias0[n] : 0.f));
                if (out1) out1[o] = f2bf(c + (bias1 ? bias1[n] : 0.f));
            }
        }
}

// ---------------------------------------------------------------------------
// Fused causal Transformer-XL relative attention (flash-style online softmax).
// Qu/Qv/Kc/Vc : bf16 [B,H,s,64] ; Rc : bf16 [H,s,64]
// outAV       : fp32 row-major [B*s, H*64]
// Grid: (s/64, B*H). 128 threads = 4 waves; wave w owns rows i0+16w..+15.
// BD via rel-shift identity: BD[i,j] = (q_i+rel_v) . r[s-1-(i-j)], j<=i.
// K tile staged by the Tensor Data Mover; R window by async global->LDS.
// ---------------------------------------------------------------------------
__global__ void __launch_bounds__(128)
relattn_wmma(const uint16_t* __restrict__ Qu, const uint16_t* __restrict__ Qv,
             const uint16_t* __restrict__ Kc, const uint16_t* __restrict__ Vc,
             const uint16_t* __restrict__ Rc, float* __restrict__ outAV,
             int s, int H)
{
    __shared__ uint16_t sK [64 * 64];       // K[j][d]   (TDM destination)
    __shared__ uint16_t sVt[64 * 64];       // V^T[d][j]
    __shared__ uint16_t sR [128 * 64];      // R window [mm][d] (async dest)
    __shared__ uint16_t sG [4][16 * 128];   // per-wave G block (bf16)
    __shared__ uint16_t sP [4][16 * 64];    // per-wave softmax probs (bf16)

    const int t    = threadIdx.x;
    const int w    = t >> 5;
    const int lane = t & 31;
    const int half = lane >> 4;
    const int l15  = lane & 15;
    const int i0   = blockIdx.x * 64;
    const int bh   = blockIdx.y;
    const int b    = bh / H;
    const int h    = bh % H;

    // Q strips (16 rows x 64 K) in A layout, loaded straight from global.
    ABf qu0, qu1, qv0, qv1;
    {
        const uint4* p = (const uint4*)(Qu + ((size_t)bh * s + i0 + w * 16 + l15) * 64);
        qu0.u[0] = p[half];     qu0.u[1] = p[2 + half];
        qu1.u[0] = p[4 + half]; qu1.u[1] = p[6 + half];
        const uint4* q = (const uint4*)(Qv + ((size_t)bh * s + i0 + w * 16 + l15) * 64);
        qv0.u[0] = q[half];     qv0.u[1] = q[2 + half];
        qv1.u[0] = q[4 + half]; qv1.u[1] = q[6 + half];
    }

    CF O[4];
    float mrow[8], lrow[8];
#pragma unroll
    for (int nt = 0; nt < 4; ++nt)
#pragma unroll
        for (int v = 0; v < 8; ++v) O[nt].f[v] = 0.f;
#pragma unroll
    for (int v = 0; v < 8; ++v) { mrow[v] = -1e30f; lrow[v] = 0.f; }

    const uint32_t sK_lds = (uint32_t)(uintptr_t)(&sK[0]);

    const int njb = blockIdx.x + 1;     // causal: key blocks j0 <= i0
    for (int jb = 0; jb < njb; ++jb) {
        const int j0 = jb * 64;

        // ---- TDM: DMA the contiguous 64x64 bf16 K tile into LDS ----
        if (w == 0) {
            uint64_t ga = (uint64_t)(uintptr_t)(Kc + ((size_t)bh * s + j0) * 64);
            tdm_load_tile_bf16(sK_lds, ga, 64u, 64u);
        }

        // ---- async global->LDS: R window (one row per thread, 8 x b128) ----
        {
            int mbase = s - 64 - i0 + j0;           // >= 0 for causal blocks
            int mr = mbase + t;                     // one R row per thread
            if (mr > s - 1) mr = s - 1;
            uint32_t goff = (uint32_t)((((size_t)h * s + mr) * 64) * 2);
            uint32_t laddr = (uint32_t)(uintptr_t)(sR + t * 64);
#pragma unroll
            for (int i = 0; i < 8; ++i)
                async_ld_b128(laddr + i * 16u, goff + i * 16u, (const void*)Rc);
        }

        // ---- V^T tile (manual transpose while copying) ----
        {
            const uint16_t* vr = Vc + ((size_t)bh * s + j0 + (t >> 1)) * 64 + (t & 1) * 32;
            uint16_t* dc = sVt + (t & 1) * 32 * 64 + (t >> 1);
#pragma unroll
            for (int dd = 0; dd < 32; ++dd) dc[dd * 64] = vr[dd];
        }
        if (jb + 1 < njb)  // prefetch next V block
            __builtin_prefetch(Vc + ((size_t)bh * s + j0 + 64 + (t >> 1)) * 64 + (t & 1) * 32, 0, 1);

        asm volatile("s_wait_asynccnt 0x0" ::: "memory");
        if (w == 0) __builtin_amdgcn_s_wait_tensorcnt(0);
        __syncthreads();

        // ---- AC = (q+u) K^T : 16x64 strip ----
        CF S[4];
#pragma unroll
        for (int nt = 0; nt < 4; ++nt) {
            ABf b0, b1;
            const uint4* pb = (const uint4*)(sK + (nt * 16 + l15) * 64);
            b0.u[0] = pb[half];     b0.u[1] = pb[2 + half];     // d 0..31
            b1.u[0] = pb[4 + half]; b1.u[1] = pb[6 + half]; // d 32..63
            v8f c = {0.f,0.f,0.f,0.f,0.f,0.f,0.f,0.f};
            c = WMMA_BF16(qu0.v, b0.v, c);
            c = WMMA_BF16(qu1.v, b1.v, c);
            S[nt].v = c;
        }

        // ---- G = (q+v) Rwin^T : 16x128, dumped to LDS for diagonal gather ----
#pragma unroll
        for (int gt = 0; gt < 8; ++gt) {
            ABf b0, b1;
            const uint4* pb = (const uint4*)(sR + (gt * 16 + l15) * 64);
            b0.u[0] = pb[half];     b0.u[1] = pb[2 + half];
            b1.u[0] = pb[4 + half]; b1.u[1] = pb[6 + half];
            v8f g = {0.f,0.f,0.f,0.f,0.f,0.f,0.f,0.f};
            g = WMMA_BF16(qv0.v, b0.v, g);
            g = WMMA_BF16(qv1.v, b1.v, g);
            CF gc; gc.v = g;
#pragma unroll
            for (int v = 0; v < 8; ++v)
                sG[w][(v + 8 * half) * 128 + gt * 16 + l15] = f2bf(gc.f[v]);
        }
        asm volatile("s_wait_dscnt 0x0" ::: "memory");   // same-wave LDS RAW

        // ---- scores + online softmax (row r lives in one 16-lane half) ----
#pragma unroll
        for (int v = 0; v < 8; ++v) {
            const int r    = v + 8 * half;
            const int irow = i0 + w * 16 + r;
            float sv[4];
#pragma unroll
            for (int nt = 0; nt < 4; ++nt) {
                int jj = nt * 16 + l15;
                float bd = bf2f(sG[w][r * 128 + (63 - (w * 16 + r) + jj)]);
                sv[nt] = (j0 + jj <= irow) ? (S[nt].f[v] + bd) * 0.125f : -1e30f;
            }
            float mx = fmaxf(fmaxf(sv[0], sv[1]), fmaxf(sv[2], sv[3]));
#pragma unroll
            for (int off = 1; off < 16; off <<= 1)
                mx = fmaxf(mx, __shfl_xor(mx, off, 32));
            float mnew = fmaxf(mrow[v], mx);
            float corr = __expf(mrow[v] - mnew);
            mrow[v] = mnew;
            float p[4], ps = 0.f;
#pragma unroll
            for (int nt = 0; nt < 4; ++nt) { p[nt] = __expf(sv[nt] - mnew); ps += p[nt]; }
#pragma unroll
            for (int off = 1; off < 16; off <<= 1) ps += __shfl_xor(ps, off, 32);
            lrow[v] = lrow[v] * corr + ps;
#pragma unroll
            for (int nt = 0; nt < 4; ++nt) O[nt].f[v] *= corr;
#pragma unroll
            for (int nt = 0; nt < 4; ++nt)
                sP[w][r * 64 + nt * 16 + l15] = f2bf(p[nt]);
        }
        asm volatile("s_wait_dscnt 0x0" ::: "memory");

        // ---- O += P V : A = P (16x64), B = V^T tiles ----
        ABf p0, p1;
        const uint4* pp = (const uint4*)(sP[w] + l15 * 64);
        p0.u[0] = pp[half];     p0.u[1] = pp[2 + half];
        p1.u[0] = pp[4 + half]; p1.u[1] = pp[6 + half];
#pragma unroll
        for (int nt = 0; nt < 4; ++nt) {
            ABf b0, b1;
            const uint4* pb = (const uint4*)(sVt + (nt * 16 + l15) * 64);
            b0.u[0] = pb[half];     b0.u[1] = pb[2 + half];     // j 0..31
            b1.u[0] = pb[4 + half]; b1.u[1] = pb[6 + half]; // j 32..63
            O[nt].v = WMMA_BF16(p0.v, b0.v, O[nt].v);
            O[nt].v = WMMA_BF16(p1.v, b1.v, O[nt].v);
        }
        __syncthreads();   // before next stage overwrites sK/sR/sVt
    }

    // ---- normalize and write attn_vec (fp32, row-major [B*s, H*64]) ----
#pragma unroll
    for (int v = 0; v < 8; ++v) {
        int r = v + 8 * half;
        float inv = 1.0f / lrow[v];
        size_t row = (size_t)(b * s + i0 + w * 16 + r);
#pragma unroll
        for (int nt = 0; nt < 4; ++nt)
            outAV[row * (size_t)(H * 64) + h * 64 + nt * 16 + l15] = O[nt].f[v] * inv;
    }
}

// ---------------------------------------------------------------------------
// Host-side orchestration
// ---------------------------------------------------------------------------
extern "C" void kernel_launch(void* const* d_in, const int* in_sizes, int n_in,
                              void* d_out, int out_size, void* d_ws, size_t ws_size,
                              hipStream_t stream)
{
    const float* q_in  = (const float*)d_in[0];
    const float* k_in  = (const float*)d_in[1];
    const float* v_in  = (const float*)d_in[2];
    const float* rel_r = (const float*)d_in[3];
    const float* rel_u = (const float*)d_in[4];   // [8,64] -> flat 512 == bias[n]
    const float* rel_v = (const float*)d_in[5];
    // d_in[6] = attn_mask (causal) -- applied analytically
    const float* Wq    = (const float*)d_in[7];
    const float* Wk    = (const float*)d_in[8];
    const float* Wv    = (const float*)d_in[9];
    const float* Wr    = (const float*)d_in[10];
    const float* fc_w  = (const float*)d_in[11];
    const float* fc_b  = (const float*)d_in[12];

    const int B = 2, S = 2048, D = 512, H = 8;
    const int M = B * S;                       // 4096

    // workspace carve-up (bf16 buffers then fp32 attn_vec) ~26 MB
    uint16_t* Qu = (uint16_t*)d_ws;
    uint16_t* Qv = Qu + (size_t)M * D;
    uint16_t* Kh = Qv + (size_t)M * D;
    uint16_t* Vh = Kh + (size_t)M * D;
    uint16_t* Rh = Vh + (size_t)M * D;         // [H, S, 64] = S*D elems
    float*    AV = (float*)(Rh + (size_t)S * D);

    dim3 blk(128);
    dim3 gP(M / 64, D / 64);                   // 64x8
    dim3 gR(S / 64, D / 64);                   // 32x8

    // projections: Q(+u -> Qu, +v -> Qv), K, V, R   (head-major bf16)
    gemm_bf16_wmma<<<gP, blk, 0, stream>>>(q_in, Wq, M, D, D,
                                           nullptr, nullptr, Qu, rel_u, Qv, rel_v, S);
    gemm_bf16_wmma<<<gP, blk, 0, stream>>>(k_in, Wk, M, D, D,
                                           nullptr, nullptr, Kh, nullptr, nullptr, nullptr, S);
    gemm_bf16_wmma<<<gP, blk, 0, stream>>>(v_in, Wv, M, D, D,
                                           nullptr, nullptr, Vh, nullptr, nullptr, nullptr, S);
    gemm_bf16_wmma<<<gR, blk, 0, stream>>>(rel_r, Wr, S, D, D,
                                           nullptr, nullptr, Rh, nullptr, nullptr, nullptr, S);

    // fused causal rel-attention -> attn_vec fp32 [M, 512]
    relattn_wmma<<<dim3(S / 64, B * H), blk, 0, stream>>>(Qu, Qv, Kh, Vh, Rh, AV, S, H);

    // final projection: out = AV @ fc_w^T + fc_b   (fp32 out)
    gemm_bf16_wmma<<<gP, blk, 0, stream>>>(AV, fc_w, M, D, D,
                                           (float*)d_out, fc_b, nullptr, nullptr, nullptr, nullptr, S);
}
